// MessagePassingLayer_35613868819191
// MI455X (gfx1250) — matrix operand order repaired and checked by reading, not compiled, not run on hardware
//
#include <hip/hip_runtime.h>
#include <hip/hip_bf16.h>

// ---------------------------------------------------------------------------
// GNN message-passing layer for MI455X (gfx1250, wave32, WMMA).
// D = 64 fixed. Both MLPs are [rows x 128] @ [128 x 64] gather-GEMMs done with
// v_wmma_f32_16x16x32_f16: 4 k-steps (K=128) x 4 n-tiles (N=64) per 16-row tile.
// All device-side indexing is unsigned 32-bit so global ops use SADDR+voffset;
// scatter atomics are emitted directly as GVS-form global_atomic_add_f32.
// ---------------------------------------------------------------------------

typedef __attribute__((ext_vector_type(16))) _Float16 v16h;
typedef __attribute__((ext_vector_type(8)))  _Float16 v8h;
typedef __attribute__((ext_vector_type(8)))  float    v8f;

__device__ inline v16h combine16(v8h lo, v8h hi) {
    v16h r;
#pragma unroll
    for (int j = 0; j < 8; ++j) { r[j] = lo[j]; r[j + 8] = hi[j]; }
    return r;
}

// Device-scope float atomic add, GVS addressing: SGPR base + 32-bit VGPR byte
// offset. Avoids per-lane 64-bit address construction in the hot loop.
__device__ inline void atomic_add_f32_gvs(float* __restrict__ base,
                                          unsigned byteOff, float val) {
    asm volatile("global_atomic_add_f32 %0, %1, %2 scope:SCOPE_DEV"
                 :: "v"(byteOff), "v"(val), "s"(base)
                 : "memory");
}

// ---------------------------------------------------------------------------
// Kernel 1: convert node features + weights to f16, zero accumulator + counts.
// ---------------------------------------------------------------------------
__global__ void init_kernel(const float* __restrict__ X,
                            const float* __restrict__ Wm,
                            const float* __restrict__ Wu,
                            _Float16* __restrict__ Xh,
                            _Float16* __restrict__ Wmh,
                            _Float16* __restrict__ Wuh,
                            float* __restrict__ acc,
                            float* __restrict__ cnt,
                            unsigned nXD, unsigned N)
{
    unsigned i = blockIdx.x * blockDim.x + threadIdx.x;
    unsigned stride = gridDim.x * blockDim.x;
    for (unsigned k = i; k < nXD; k += stride) {
        Xh[k]  = (_Float16)X[k];
        acc[k] = 0.0f;
    }
    for (unsigned k = i; k < 64u * 128u; k += stride) {
        Wmh[k] = (_Float16)Wm[k];
        Wuh[k] = (_Float16)Wu[k];
    }
    for (unsigned k = i; k < N; k += stride) cnt[k] = 0.0f;
}

// ---------------------------------------------------------------------------
// Kernel 2: per-edge message MLP + scatter-add aggregation (full 16-edge tiles).
// Weights (16 B-fragments) preloaded to registers; grid-stride over tiles.
// ---------------------------------------------------------------------------
__global__ __launch_bounds__(256) void edge_msg_kernel(
    const int* __restrict__ edge_index,       // [E][2] (src, tgt)
    const _Float16* __restrict__ Xh,          // [N][64]
    const _Float16* __restrict__ Wmh,         // [64][128] (B[k][n] = Wmh[n*128+k])
    const float* __restrict__ b_msg,          // [64]
    float* __restrict__ acc,                  // [N][64]
    float* __restrict__ cnt,                  // [N]
    unsigned numTiles)
{
    const unsigned lane = threadIdx.x & 31u;
    const unsigned wave = threadIdx.x >> 5;
    const unsigned row  = lane & 15u;         // local edge row / output col
    const unsigned kg   = lane >> 4;          // k half-group

    // ---- Preload all 16 weight fragments + biases (held for whole kernel) ----
    v16h B[4][4];
    float bias[4];
    {
        const v8h* wb = (const v8h*)Wmh;
#pragma unroll
        for (int t = 0; t < 4; ++t) {
            const unsigned n = (unsigned)t * 16u + row;
            const unsigned base = n * 16u + kg;          // v8h chunks
#pragma unroll
            for (int s = 0; s < 4; ++s)
                B[t][s] = combine16(wb[base + 4u * s], wb[base + 4u * s + 2u]);
            bias[t] = b_msg[n];
        }
    }

    const v8h*  xb = (const v8h*)Xh;
    const int2* e2 = (const int2*)edge_index;
    const int4* e4 = (const int4*)edge_index;

    for (unsigned tile = blockIdx.x * 8u + wave; tile < numTiles;
         tile += gridDim.x * 8u) {
        const unsigned e0 = tile * 16u;

        // own edge (for A gather + count)
        const int2 st = e2[e0 + row];
        const unsigned sb = (unsigned)st.x * 8u;   // v8h chunks per feature row
        const unsigned tb = (unsigned)st.y * 8u;

        // A fragments: [x_src | x_tgt], 16x128 f16
        v16h a0 = combine16(xb[sb + kg],      xb[sb + kg + 2u]);   // K   0..31
        v16h a1 = combine16(xb[sb + kg + 4u], xb[sb + kg + 6u]);   // K  32..63
        v16h a2 = combine16(xb[tb + kg],      xb[tb + kg + 2u]);   // K  64..95
        v16h a3 = combine16(xb[tb + kg + 4u], xb[tb + kg + 6u]);   // K  96..127

        // byte offsets of acc rows for the 8 C-rows this lane owns (M = kg*8+i)
        unsigned accOffB[8];
        {
            const unsigned qb = tile * 8u + kg * 4u;   // int4 index (2 edges each)
#pragma unroll
            for (int q = 0; q < 4; ++q) {
                const int4 p = e4[qb + q];
                accOffB[2 * q]     = (unsigned)p.y * 256u;   // 64 floats * 4B
                accOffB[2 * q + 1] = (unsigned)p.w * 256u;
            }
        }

#pragma unroll
        for (int t = 0; t < 4; ++t) {
            v8f c = {};
            c = __builtin_amdgcn_wmma_f32_16x16x32_f16(false, a0, false, B[t][0], (short)0, c, false, false);
            c = __builtin_amdgcn_wmma_f32_16x16x32_f16(false, a1, false, B[t][1], (short)0, c, false, false);
            c = __builtin_amdgcn_wmma_f32_16x16x32_f16(false, a2, false, B[t][2], (short)0, c, false, false);
            c = __builtin_amdgcn_wmma_f32_16x16x32_f16(false, a3, false, B[t][3], (short)0, c, false, false);

            const unsigned n4 = ((unsigned)t * 16u + row) * 4u;  // feature byte off
#pragma unroll
            for (int i = 0; i < 8; ++i) {
                float v = c[i] + bias[t];
                v = v > 0.0f ? v : 0.0f;               // relu
                atomic_add_f32_gvs(acc, accOffB[i] + n4, v);
            }
        }
        if (kg == 0u) atomic_add_f32_gvs(cnt, (unsigned)st.y * 4u, 1.0f);
    }
}

// ---------------------------------------------------------------------------
// Kernel 2b: scalar fallback for remainder edges (E % 16), exact fp32 math.
// ---------------------------------------------------------------------------
__global__ void edge_remainder_kernel(
    const int* __restrict__ edge_index,
    const float* __restrict__ X,
    const float* __restrict__ Wm,             // [64][128]
    const float* __restrict__ b_msg,
    float* __restrict__ acc,
    float* __restrict__ cnt,
    unsigned start, unsigned E)
{
    const unsigned n = threadIdx.x;            // 64 threads, one per feature
    for (unsigned e = start; e < E; ++e) {
        const unsigned src = (unsigned)edge_index[2 * e];
        const unsigned tgt = (unsigned)edge_index[2 * e + 1];
        float s = b_msg[n];
        for (unsigned k = 0; k < 64u; ++k) {
            s += X[src * 64u + k] * Wm[n * 128u + k];
            s += X[tgt * 64u + k] * Wm[n * 128u + 64u + k];
        }
        s = s > 0.0f ? s : 0.0f;
        atomicAdd(acc + tgt * 64u + n, s);
        if (n == 0u) atomicAdd(cnt + tgt, 1.0f);
    }
}

// ---------------------------------------------------------------------------
// Kernel 3: node update MLP + residual + LayerNorm (16 nodes per wave).
// ---------------------------------------------------------------------------
__global__ __launch_bounds__(256) void update_kernel(
    const float* __restrict__ X,              // [N][64] fp32 (residual)
    const _Float16* __restrict__ Xh,          // [N][64]
    const _Float16* __restrict__ Wuh,         // [64][128]
    const float* __restrict__ b_upd,          // [64]
    const float* __restrict__ gamma,          // [64]
    const float* __restrict__ beta,           // [64]
    const float* __restrict__ acc,            // [N][64]
    const float* __restrict__ cnt,            // [N]
    float* __restrict__ out,                  // [N][64]
    unsigned numTiles, unsigned N)
{
    const unsigned lane = threadIdx.x & 31u;
    const unsigned wave = threadIdx.x >> 5;
    const unsigned row  = lane & 15u;
    const unsigned kg   = lane >> 4;
    const unsigned tile = blockIdx.x * 8u + wave;   // wave-uniform
    if (tile >= numTiles) return;

    const unsigned node  = tile * 16u + row;
    const unsigned nodeC = (node < N) ? node : (N - 1u);

    // --- A fragments: x (k=0..63), msg_mean (k=64..127) ---
    const v8h* xb = (const v8h*)Xh;
    const unsigned nb = nodeC * 8u;
    v16h a0 = combine16(xb[nb + kg],      xb[nb + kg + 2u]);
    v16h a1 = combine16(xb[nb + kg + 4u], xb[nb + kg + 6u]);

    const float inv = 1.0f / fmaxf(cnt[nodeC], 1.0f);
    const float* pm = acc + nodeC * 64u;
    v16h a2, a3;
#pragma unroll
    for (int j = 0; j < 8; ++j) {
        a2[j]     = (_Float16)(pm[kg * 8u + j]       * inv);
        a2[j + 8] = (_Float16)(pm[16u + kg * 8u + j] * inv);
        a3[j]     = (_Float16)(pm[32u + kg * 8u + j] * inv);
        a3[j + 8] = (_Float16)(pm[48u + kg * 8u + j] * inv);
    }

    float g[4], be[4];
#pragma unroll
    for (int t = 0; t < 4; ++t) {
        g[t]  = gamma[(unsigned)t * 16u + row];
        be[t] = beta[(unsigned)t * 16u + row];
    }

    v8f y[4];
    const v8h* wb = (const v8h*)Wuh;
#pragma unroll
    for (int t = 0; t < 4; ++t) {
        const unsigned n = (unsigned)t * 16u + row;
        const unsigned base = n * 16u + kg;
        v16h b0 = combine16(wb[base],       wb[base + 2u]);
        v16h b1 = combine16(wb[base + 4u],  wb[base + 6u]);
        v16h b2 = combine16(wb[base + 8u],  wb[base + 10u]);
        v16h b3 = combine16(wb[base + 12u], wb[base + 14u]);

        v8f c = {};
        c = __builtin_amdgcn_wmma_f32_16x16x32_f16(false, a0, false, b0, (short)0, c, false, false);
        c = __builtin_amdgcn_wmma_f32_16x16x32_f16(false, a1, false, b1, (short)0, c, false, false);
        c = __builtin_amdgcn_wmma_f32_16x16x32_f16(false, a2, false, b2, (short)0, c, false, false);
        c = __builtin_amdgcn_wmma_f32_16x16x32_f16(false, a3, false, b3, (short)0, c, false, false);

        const float bias = b_upd[n];
#pragma unroll
        for (int i = 0; i < 8; ++i) {
            const unsigned nm  = tile * 16u + kg * 8u + i;   // C row M = kg*8 + i
            const unsigned nmC = (nm < N) ? nm : (N - 1u);
            float v = c[i] + bias;
            v = v > 0.0f ? v : 0.0f;                         // relu
            y[t][i] = v + X[nmC * 64u + n];                  // residual
        }
    }

    // --- LayerNorm across the 64 features of each row ---
#pragma unroll
    for (int i = 0; i < 8; ++i) {
        float s1 = y[0][i] + y[1][i] + y[2][i] + y[3][i];
        float s2 = y[0][i] * y[0][i] + y[1][i] * y[1][i]
                 + y[2][i] * y[2][i] + y[3][i] * y[3][i];
#pragma unroll
        for (int mask = 1; mask < 16; mask <<= 1) {          // reduce 16-lane group
            s1 += __shfl_xor(s1, mask, 32);
            s2 += __shfl_xor(s2, mask, 32);
        }
        const float mu  = s1 * (1.0f / 64.0f);
        const float var = s2 * (1.0f / 64.0f) - mu * mu;
        const float rs  = rsqrtf(var + 1e-5f);
        const unsigned nm = tile * 16u + kg * 8u + i;
        if (nm < N) {
#pragma unroll
            for (int t = 0; t < 4; ++t) {
                const unsigned n = (unsigned)t * 16u + row;
                out[nm * 64u + n] = (y[t][i] - mu) * rs * g[t] + be[t];
            }
        }
    }
}

// ---------------------------------------------------------------------------
extern "C" void kernel_launch(void* const* d_in, const int* in_sizes, int n_in,
                              void* d_out, int out_size, void* d_ws, size_t ws_size,
                              hipStream_t stream) {
    const float* X     = (const float*)d_in[0];
    const int*   EI    = (const int*)d_in[1];
    const float* Wm    = (const float*)d_in[2];
    const float* bm    = (const float*)d_in[3];
    const float* Wu    = (const float*)d_in[4];
    const float* bu    = (const float*)d_in[5];
    const float* gamma = (const float*)d_in[6];
    const float* beta  = (const float*)d_in[7];
    float* outp = (float*)d_out;

    const unsigned N   = (unsigned)(in_sizes[0] / 64);
    const unsigned E   = (unsigned)(in_sizes[1] / 2);
    const unsigned nXD = N * 64u;

    // Workspace layout (256B aligned)
    char* ws = (char*)d_ws;
    size_t off = 0;
    auto take = [&](size_t bytes) { size_t o = off; off += (bytes + 255) & ~(size_t)255; return o; };
    _Float16* Xh  = (_Float16*)(ws + take((size_t)nXD * sizeof(_Float16)));
    _Float16* Wmh = (_Float16*)(ws + take(64 * 128 * sizeof(_Float16)));
    _Float16* Wuh = (_Float16*)(ws + take(64 * 128 * sizeof(_Float16)));
    float*    acc = (float*)   (ws + take((size_t)nXD * sizeof(float)));
    float*    cnt = (float*)   (ws + take((size_t)N * sizeof(float)));

    // 1) convert + zero
    {
        unsigned blocks = (nXD + 255u) / 256u;
        if (blocks > 16384u) blocks = 16384u;
        init_kernel<<<blocks, 256, 0, stream>>>(X, Wm, Wu, Xh, Wmh, Wuh, acc, cnt, nXD, N);
    }
    // 2) per-edge messages + aggregation (16 edges per wave, grid-stride)
    {
        const unsigned numTiles = E / 16u;               // full tiles only
        if (numTiles > 0u) {
            unsigned blocks = (numTiles + 7u) / 8u;
            if (blocks > 4096u) blocks = 4096u;
            edge_msg_kernel<<<blocks, 256, 0, stream>>>(EI, Xh, Wmh, bm, acc, cnt, numTiles);
        }
        if (E % 16u)                                      // remainder edges (scalar)
            edge_remainder_kernel<<<1, 64, 0, stream>>>(EI, X, Wm, bm, acc, cnt,
                                                        numTiles * 16u, E);
    }
    // 3) node update + residual + layernorm (16 nodes per wave)
    {
        const unsigned numTiles = (N + 15u) / 16u;
        const unsigned blocks   = (numTiles + 7u) / 8u;
        update_kernel<<<blocks, 256, 0, stream>>>(X, Xh, Wuh, bu, gamma, beta,
                                                  acc, cnt, outp, numTiles, N);
    }
}
